// DeepSAT_20383914787532
// MI455X (gfx1250) — matrix operand order repaired and checked by reading, not compiled, not run on hardware
//
#include <hip/hip_runtime.h>
#include <hip/hip_bf16.h>
#include <math.h>

typedef __attribute__((ext_vector_type(8)))  float  v8f;
typedef __attribute__((ext_vector_type(16))) __bf16 v16bf;
typedef __attribute__((ext_vector_type(8)))  __bf16 v8bf;

#define DH   128
#define DX   16
#define DM   32
#define LMAX 16
#define KIH  160          // gi K (144) padded to 160
#define KA   288          // LDS A row: msg(128) | x(16) | zero-pad(16) | h(128)

union AFrag { v16bf v; v8bf h2[2]; };

__device__ __forceinline__ v8f wmma_bf16(v16bf a, v16bf b, v8f c) {
    return __builtin_amdgcn_wmma_f32_16x16x32_bf16(false, a, false, b, (short)0, c, false, false);
}
__device__ __forceinline__ float sigmoidf_(float x) { return 1.0f / (1.0f + __expf(-x)); }

// ---------------- weight prep (fp32 -> bf16, B-fragment layout [n][k]) ----------------
__global__ void k_prep_weights(const float* agg_w, const float* w_ih, const float* w_hh,
                               const float* m1, const float* m2,
                               __bf16* Bagg, __bf16* Bih, __bf16* Bhh,
                               __bf16* Bm1, __bf16* Bm2) {
    int i = blockIdx.x * blockDim.x + threadIdx.x;
    if (i < DH * DH)      Bagg[i] = (__bf16)agg_w[i];
    if (i < 3 * DH * KIH) {
        int n = i / KIH, k = i % KIH;
        Bih[i] = (k < DH + DX) ? (__bf16)w_ih[n * (DH + DX) + k] : (__bf16)0.0f;
    }
    if (i < 3 * DH * DH)  Bhh[i] = (__bf16)w_hh[i];
    if (i < DM * DH)      Bm1[i] = (__bf16)m1[i];
    if (i < DM * DM)      Bm2[i] = (__bf16)m2[i];
}

// t0[d] = sum_k agg_w[d][k]*h0[k] + agg_b[d] ; h0 = emd_w + emd_b (one row, broadcast later)
__global__ void k_t0(const float* emd_w, const float* emd_b,
                     const float* agg_w, const float* agg_b, float* h0, float* t0) {
    int d = threadIdx.x;
    h0[d] = emd_w[d] + emd_b[d];
    float s = agg_b[d];
    for (int k = 0; k < DH; ++k) s += agg_w[d * DH + k] * (emd_w[k] + emd_b[k]);
    t0[d] = s;
}

__global__ void k_broadcast(const float* h0, const float* t0, float* h, float* t, int N) {
    long gid = (long)blockIdx.x * blockDim.x + threadIdx.x;
    if (gid >= (long)N * 32) return;
    int c = (int)(gid & 31) * 4;
    long n = gid >> 5;
    float4 hv = *(const float4*)(h0 + c);
    float4 tv = *(const float4*)(t0 + c);
    *(float4*)(h + n * DH + c) = hv;
    *(float4*)(t + n * DH + c) = tv;
}

// ---------------- level bucketing (static over the run) ----------------
__global__ void k_zero_counts(int* cnts) { if (threadIdx.x < 64) cnts[threadIdx.x] = 0; }

__global__ void k_count(const int* level, int N, const int* tgt, int E,
                        int* node_cnt, int* edge_cnt) {
    int i = blockIdx.x * blockDim.x + threadIdx.x;
    if (i < N) atomicAdd(&node_cnt[level[i] & 15], 1);
    if (i < E) atomicAdd(&edge_cnt[level[tgt[i]] & 15], 1);
}

__global__ void k_prefix(const int* node_cnt, const int* edge_cnt, int* node_off, int* edge_off) {
    if (threadIdx.x == 0) {
        int a = 0, b = 0;
        for (int l = 0; l < 16; ++l) {
            node_off[l] = a; a += node_cnt[l];
            edge_off[l] = b; b += edge_cnt[l];
        }
    }
}

__global__ void k_fill(const int* level, int N, const int* tgt, int E,
                       const int* node_off, const int* edge_off,
                       int* node_fill, int* edge_fill, int* lvl_nodes, int* lvl_edges) {
    int i = blockIdx.x * blockDim.x + threadIdx.x;
    if (i < N) {
        int lv = level[i] & 15;
        int p = atomicAdd(&node_fill[lv], 1);
        lvl_nodes[node_off[lv] + p] = i;
    }
    if (i < E) {
        int lv = level[tgt[i]] & 15;
        int p = atomicAdd(&edge_fill[lv], 1);
        lvl_edges[edge_off[lv] + p] = i;
    }
}

// ---------------- per-level sparse pipeline ----------------
__global__ void k_zero_msg(int l, const int* lvl_nodes, const int* node_off,
                           const int* node_cnt, float* msg) {
    long gid = (long)blockIdx.x * blockDim.x + threadIdx.x;
    int ni = (int)(gid >> 5);
    if (ni >= node_cnt[l]) return;
    int node = lvl_nodes[node_off[l] + ni];
    int c = (int)(gid & 31) * 4;
    float4 z = {0.f, 0.f, 0.f, 0.f};
    *(float4*)(msg + (long)node * DH + c) = z;
}

__global__ void k_scatter(int l, const int* lvl_edges, const int* edge_off, const int* edge_cnt,
                          const int* src, const int* tgt, const float* t, float* msg) {
    long gid = (long)blockIdx.x * blockDim.x + threadIdx.x;
    int ei = (int)(gid >> 5);
    if (ei >= edge_cnt[l]) return;
    int e = lvl_edges[edge_off[l] + ei];
    int s = src[e], d = tgt[e];
    int c = (int)(gid & 31) * 4;
    const float4 v = *(const float4*)(t + (long)s * DH + c);
    float* mp = msg + (long)d * DH + c;
    atomicAdd(mp + 0, v.x);
    atomicAdd(mp + 1, v.y);
    atomicAdd(mp + 2, v.z);
    atomicAdd(mp + 3, v.w);
}

// Fused: GRU update for compacted level-l nodes + incremental refresh of t = agg(h).
// One wave per 16-row tile; 248 v_wmma_f32_16x16x32_bf16 per tile.
// A fragments are preloaded into register arrays; B fragments are batch-loaded per
// column tile so the WMMA chain runs back-to-back after a single wait.
__launch_bounds__(32)
__global__ void k_gru_level(int l, const int* lvl_nodes, const int* node_off, const int* node_cnt,
                            const float* msg, const float* x, float* h, float* t,
                            const __bf16* Bih, const __bf16* Bhh, const __bf16* Bagg,
                            const float* b_ih, const float* b_hh, const float* agg_b) {
    __shared__ __bf16 Als[16 * KA];        // 9216 B
    __shared__ float  Gi[16 * 3 * DH];     // 24576 B
    __shared__ float  Gh[16 * 3 * DH];     // 24576 B
    __shared__ __bf16 Hn[16 * DH];         // 4096 B
    __shared__ int    nodes[16];
    __shared__ int    act[16];

    int cnt = node_cnt[l];
    int tile = blockIdx.x;
    if (tile * 16 >= cnt) return;
    int tid = threadIdx.x;
    int lane = tid & 31, lh = lane >> 4, lm = lane & 15;

    if (tid < 16) {
        int ridx = tile * 16 + tid;
        int a = ridx < cnt;
        int rr = a ? ridx : cnt - 1;
        nodes[tid] = lvl_nodes[node_off[l] + rr];
        act[tid] = a;
    }
    __syncthreads();

    // Stage A rows into LDS as bf16: [msg | x | 0 | h]
    for (int j = tid; j < 16 * KA; j += 32) {
        int r = j / KA, c = j % KA;
        int node = nodes[r];
        float v;
        if (c < DH)                 v = msg[(long)node * DH + c];
        else if (c < DH + DX)       v = x[(long)node * DX + (c - DH)];
        else if (c < DH + DX + 16)  v = 0.0f;
        else                        v = h[(long)node * DH + (c - (DH + DX + 16))];
        Als[j] = (__bf16)v;
    }
    __syncthreads();

    // ---- gi = [msg|x] @ w_ih^T + b_ih  (K = 160 padded, 24 col tiles of 16) ----
    {
        AFrag ai[5];
        #pragma unroll
        for (int ks = 0; ks < 5; ++ks) {
            ai[ks].h2[0] = *(const v8bf*)&Als[lm * KA + ks * 32 + lh * 16];
            ai[ks].h2[1] = *(const v8bf*)&Als[lm * KA + ks * 32 + lh * 16 + 8];
        }
        for (int nt = 0; nt < 24; ++nt) {
            v16bf b[5];
            #pragma unroll
            for (int ks = 0; ks < 5; ++ks)
                b[ks] = *(const v16bf*)&Bih[(nt * 16 + lm) * KIH + ks * 32 + lh * 16];
            v8f acc = {};
            #pragma unroll
            for (int ks = 0; ks < 5; ++ks) acc = wmma_bf16(ai[ks].v, b[ks], acc);
            int n = nt * 16 + lm;
            float bias = b_ih[n];
            #pragma unroll
            for (int r = 0; r < 8; ++r)
                Gi[(r + lh * 8) * (3 * DH) + n] = acc[r] + bias;
        }
    }
    // ---- gh = h @ w_hh^T + b_hh  (A columns 160..287 of LDS) ----
    {
        AFrag ah[4];
        #pragma unroll
        for (int ks = 0; ks < 4; ++ks) {
            int kb = 160 + ks * 32;
            ah[ks].h2[0] = *(const v8bf*)&Als[lm * KA + kb + lh * 16];
            ah[ks].h2[1] = *(const v8bf*)&Als[lm * KA + kb + lh * 16 + 8];
        }
        for (int nt = 0; nt < 24; ++nt) {
            v16bf b[4];
            #pragma unroll
            for (int ks = 0; ks < 4; ++ks)
                b[ks] = *(const v16bf*)&Bhh[(nt * 16 + lm) * DH + ks * 32 + lh * 16];
            v8f acc = {};
            #pragma unroll
            for (int ks = 0; ks < 4; ++ks) acc = wmma_bf16(ah[ks].v, b[ks], acc);
            int n = nt * 16 + lm;
            float bias = b_hh[n];
            #pragma unroll
            for (int r = 0; r < 8; ++r)
                Gh[(r + lh * 8) * (3 * DH) + n] = acc[r] + bias;
        }
    }
    __syncthreads();

    // Elementwise GRU gates; write h and stage new h (bf16) for the t refresh.
    for (int j = tid; j < 16 * DH; j += 32) {
        int r = j / DH, d = j % DH;
        int node = nodes[r];
        float hold = h[(long)node * DH + d];
        float hn;
        if (act[r]) {
            float ir = Gi[r * 3 * DH + d], iz = Gi[r * 3 * DH + DH + d], in_ = Gi[r * 3 * DH + 2 * DH + d];
            float hr = Gh[r * 3 * DH + d], hz = Gh[r * 3 * DH + DH + d], hnn = Gh[r * 3 * DH + 2 * DH + d];
            float rg = sigmoidf_(ir + hr);
            float zg = sigmoidf_(iz + hz);
            float ng = tanhf(in_ + rg * hnn);
            hn = (1.0f - zg) * ng + zg * hold;
            h[(long)node * DH + d] = hn;
        } else hn = hold;
        Hn[j] = (__bf16)hn;
    }
    __syncthreads();

    // t[node] = h_new @ agg_w^T + agg_b for updated rows only (incremental maintenance).
    {
        AFrag an[4];
        #pragma unroll
        for (int ks = 0; ks < 4; ++ks) {
            an[ks].h2[0] = *(const v8bf*)&Hn[lm * DH + ks * 32 + lh * 16];
            an[ks].h2[1] = *(const v8bf*)&Hn[lm * DH + ks * 32 + lh * 16 + 8];
        }
        for (int nt = 0; nt < 8; ++nt) {
            v16bf b[4];
            #pragma unroll
            for (int ks = 0; ks < 4; ++ks)
                b[ks] = *(const v16bf*)&Bagg[(nt * 16 + lm) * DH + ks * 32 + lh * 16];
            v8f acc = {};
            #pragma unroll
            for (int ks = 0; ks < 4; ++ks) acc = wmma_bf16(an[ks].v, b[ks], acc);
            int n = nt * 16 + lm;
            float bias = agg_b[n];
            #pragma unroll
            for (int r = 0; r < 8; ++r) {
                int m = r + lh * 8;
                if (act[m]) t[(long)nodes[m] * DH + n] = acc[r] + bias;
            }
        }
    }
}

// ---------------- fused 3-layer MLP head ----------------
__launch_bounds__(256)
__global__ void k_mlp(const float* h, const __bf16* Bm1, const __bf16* Bm2,
                      const float* b1, const float* b2, const float* w3, const float* b3,
                      float* out, int N) {
    __shared__ __bf16 Z1[8][16 * DM];   // 8 KB
    __shared__ float  Z2[8][16 * DM];   // 16 KB
    int wave = threadIdx.x >> 5;
    int lane = threadIdx.x & 31, lh = lane >> 4, lm = lane & 15;
    long rbase = (long)blockIdx.x * 128 + wave * 16;

    // z1 = relu(h @ m1^T + b1): K=128 (4 steps), 2 col tiles.
    // Preload all A and B fragments, then run the 8 WMMAs back-to-back.
    long row = rbase + lm; if (row >= N) row = N - 1;
    AFrag a[4];
    #pragma unroll
    for (int ks = 0; ks < 4; ++ks) {
        int kb = ks * 32;
        const float* hp  = h + row * DH + kb + lh * 16;
        const float* hp2 = h + row * DH + kb + lh * 16 + 8;
        #pragma unroll
        for (int i = 0; i < 8; ++i) { a[ks].v[i] = (__bf16)hp[i]; a[ks].v[8 + i] = (__bf16)hp2[i]; }
    }
    v16bf b0[4], b1f[4];
    #pragma unroll
    for (int ks = 0; ks < 4; ++ks) {
        b0[ks]  = *(const v16bf*)&Bm1[(0 * 16 + lm) * DH + ks * 32 + lh * 16];
        b1f[ks] = *(const v16bf*)&Bm1[(1 * 16 + lm) * DH + ks * 32 + lh * 16];
    }
    v8f acc0 = {}, acc1 = {};
    #pragma unroll
    for (int ks = 0; ks < 4; ++ks) {
        acc0 = wmma_bf16(a[ks].v, b0[ks], acc0);
        acc1 = wmma_bf16(a[ks].v, b1f[ks], acc1);
    }
    #pragma unroll
    for (int r = 0; r < 8; ++r) {
        int m = r + lh * 8;
        Z1[wave][m * DM + lm]      = (__bf16)fmaxf(acc0[r] + b1[lm], 0.f);
        Z1[wave][m * DM + 16 + lm] = (__bf16)fmaxf(acc1[r] + b1[16 + lm], 0.f);
    }
    __syncthreads();

    // z2 = relu(z1 @ m2^T + b2): K=32 (1 step), 2 col tiles
    v8f c0 = {}, c1 = {};
    {
        AFrag az;
        az.h2[0] = *(const v8bf*)&Z1[wave][lm * DM + lh * 16];
        az.h2[1] = *(const v8bf*)&Z1[wave][lm * DM + lh * 16 + 8];
        v16bf bb0 = *(const v16bf*)&Bm2[lm * DM + lh * 16];
        v16bf bb1 = *(const v16bf*)&Bm2[(16 + lm) * DM + lh * 16];
        c0 = wmma_bf16(az.v, bb0, c0);
        c1 = wmma_bf16(az.v, bb1, c1);
    }
    #pragma unroll
    for (int r = 0; r < 8; ++r) {
        int m = r + lh * 8;
        Z2[wave][m * DM + lm]      = fmaxf(c0[r] + b2[lm], 0.f);
        Z2[wave][m * DM + 16 + lm] = fmaxf(c1[r] + b2[16 + lm], 0.f);
    }
    __syncthreads();

    // final 32 -> 1 dot per row
    if (lane < 16) {
        long r = rbase + lane;
        if (r < N) {
            float s = b3[0];
            #pragma unroll
            for (int k = 0; k < DM; ++k) s += Z2[wave][lane * DM + k] * w3[k];
            out[r] = s;
        }
    }
}

// ---------------- launcher ----------------
extern "C" void kernel_launch(void* const* d_in, const int* in_sizes, int n_in,
                              void* d_out, int out_size, void* d_ws, size_t ws_size,
                              hipStream_t stream) {
    (void)n_in; (void)out_size; (void)ws_size;
    const float* x     = (const float*)d_in[0];
    const int*   eidx  = (const int*)d_in[1];
    const int*   level = (const int*)d_in[2];
    // d_in[3] = num_layers_f (device scalar; structurally L = 16)
    const float* emd_w = (const float*)d_in[4];
    const float* emd_b = (const float*)d_in[5];
    const float* agg_w = (const float*)d_in[6];
    const float* agg_b = (const float*)d_in[7];
    const float* w_ih  = (const float*)d_in[8];
    const float* w_hh  = (const float*)d_in[9];
    const float* b_ih  = (const float*)d_in[10];
    const float* b_hh  = (const float*)d_in[11];
    const float* m1    = (const float*)d_in[12];
    const float* mb1   = (const float*)d_in[13];
    const float* m2    = (const float*)d_in[14];
    const float* mb2   = (const float*)d_in[15];
    const float* m3    = (const float*)d_in[16];
    const float* mb3   = (const float*)d_in[17];

    int N = in_sizes[0] / DX;
    int E = in_sizes[1] / 2;
    const int* src = eidx;
    const int* tgt = eidx + E;

    // workspace carve-out (256B aligned)
    char* w = (char*)d_ws;
    auto alloc = [&](size_t bytes) -> char* {
        char* p = w;
        w += (bytes + 255) & ~(size_t)255;
        return p;
    };
    float*  h       = (float*)alloc((size_t)N * DH * 4);
    float*  t       = (float*)alloc((size_t)N * DH * 4);
    float*  msg     = (float*)alloc((size_t)N * DH * 4);
    float*  h0      = (float*)alloc(DH * 4);
    float*  t0      = (float*)alloc(DH * 4);
    __bf16* Bagg    = (__bf16*)alloc(DH * DH * 2);
    __bf16* Bih     = (__bf16*)alloc(3 * DH * KIH * 2);
    __bf16* Bhh     = (__bf16*)alloc(3 * DH * DH * 2);
    __bf16* Bm1     = (__bf16*)alloc(DM * DH * 2);
    __bf16* Bm2     = (__bf16*)alloc(DM * DM * 2);
    int*    cnts    = (int*)alloc(64 * 4);
    int*    node_off = (int*)alloc(16 * 4);
    int*    edge_off = (int*)alloc(16 * 4);
    int*    lvl_nodes = (int*)alloc((size_t)N * 4);
    int*    lvl_edges = (int*)alloc((size_t)E * 4);
    int *node_cnt = cnts, *edge_cnt = cnts + 16, *node_fill = cnts + 32, *edge_fill = cnts + 48;

    int mNE = (N > E) ? N : E;

    k_prep_weights<<<(3 * DH * KIH + 255) / 256, 256, 0, stream>>>(
        agg_w, w_ih, w_hh, m1, m2, Bagg, Bih, Bhh, Bm1, Bm2);
    k_t0<<<1, DH, 0, stream>>>(emd_w, emd_b, agg_w, agg_b, h0, t0);
    k_broadcast<<<(int)(((size_t)N * 32 + 255) / 256), 256, 0, stream>>>(h0, t0, h, t, N);
    k_zero_counts<<<1, 64, 0, stream>>>(cnts);
    k_count<<<(mNE + 255) / 256, 256, 0, stream>>>(level, N, tgt, E, node_cnt, edge_cnt);
    k_prefix<<<1, 1, 0, stream>>>(node_cnt, edge_cnt, node_off, edge_off);
    k_fill<<<(mNE + 255) / 256, 256, 0, stream>>>(level, N, tgt, E, node_off, edge_off,
                                                  node_fill, edge_fill, lvl_nodes, lvl_edges);

    int zeroGrid = (int)(((size_t)N * 32 + 255) / 256);
    int scatGrid = (int)(((size_t)E * 32 + 255) / 256);
    int gruGrid  = (N + 15) / 16;
    for (int l = 1; l < LMAX; ++l) {
        k_zero_msg<<<zeroGrid, 256, 0, stream>>>(l, lvl_nodes, node_off, node_cnt, msg);
        k_scatter<<<scatGrid, 256, 0, stream>>>(l, lvl_edges, edge_off, edge_cnt, src, tgt, t, msg);
        k_gru_level<<<gruGrid, 32, 0, stream>>>(l, lvl_nodes, node_off, node_cnt,
                                                msg, x, h, t, Bih, Bhh, Bagg, b_ih, b_hh, agg_b);
    }
    k_mlp<<<(N + 127) / 128, 256, 0, stream>>>(h, Bm1, Bm2, mb1, mb2, m3, mb3, (float*)d_out, N);
}
